// SNN_2_78039555768627
// MI455X (gfx1250) — matrix operand order repaired and checked by reading, not compiled, not run on hardware
//
#include <hip/hip_runtime.h>

typedef __attribute__((ext_vector_type(16))) _Float16 v16h;
typedef __attribute__((ext_vector_type(8)))  _Float16 v8h;
typedef __attribute__((ext_vector_type(8)))  float    v8f;

#define V_TH   1.0f
#define BN_EPS 1e-5f

// ---------------------------------------------------------------- utilities
__global__ void fill_zero_f32(float* p, int n) {
    int i = blockIdx.x * blockDim.x + threadIdx.x;
    if (i < n) p[i] = 0.f;
}

__global__ void fill_zero_f16(_Float16* p, int n) {
    int i = blockIdx.x * blockDim.x + threadIdx.x;
    if (i < n) p[i] = (_Float16)0.f;
}

// Pack fw1 [100,1200] f32 -> Bt [112,1216] f16 (row = output neuron, contiguous K)
__global__ void pack_fw1(const float* __restrict__ fw1, _Float16* __restrict__ Bt) {
    int i = blockIdx.x * blockDim.x + threadIdx.x;
    if (i >= 112 * 1216) return;
    int o = i / 1216, k = i - o * 1216;
    float v = (o < 100 && k < 1200) ? fw1[o * 1200 + k] : 0.f;
    Bt[i] = (_Float16)v;
}

// ------------------------------------------------- conv3x3 (pad=1) + BN stats
template<int CIN>
__global__ void conv3x3_stats(const float* __restrict__ in, const float* __restrict__ w,
                              float* __restrict__ y, float* __restrict__ stats,
                              int N, int COUT, int H, int W, int blocksPerPlane) {
    int bid   = blockIdx.x;
    int plane = bid / blocksPerPlane;        // plane = n*COUT + oc
    int chunk = bid - plane * blocksPerPlane;
    int n  = plane / COUT;
    int oc = plane - n * COUT;
    int tid = threadIdx.x;
    int pix = chunk * 256 + tid;
    int HW  = H * W;
    float acc = 0.f;
    bool active = pix < HW;
    if (active) {
        int oh = pix / W, ow = pix - oh * W;
        #pragma unroll
        for (int ic = 0; ic < CIN; ++ic) {
            const float* ip = in + ((size_t)n * CIN + ic) * HW;
            const float* wp = w  + ((size_t)oc * CIN + ic) * 9;
            #pragma unroll
            for (int kh = 0; kh < 3; ++kh) {
                int ih = oh + kh - 1;
                if (ih < 0 || ih >= H) continue;
                #pragma unroll
                for (int kw = 0; kw < 3; ++kw) {
                    int iw = ow + kw - 1;
                    if (iw < 0 || iw >= W) continue;
                    acc = fmaf(ip[ih * W + iw], wp[kh * 3 + kw], acc);
                }
            }
        }
        y[(size_t)plane * HW + pix] = acc;
    }
    // per-channel sum / sumsq partial reduction (oc uniform across block)
    __shared__ float rs[256], rq[256];
    float v = active ? acc : 0.f;
    rs[tid] = v; rq[tid] = v * v;
    __syncthreads();
    for (int s = 128; s > 0; s >>= 1) {
        if (tid < s) { rs[tid] += rs[tid + s]; rq[tid] += rq[tid + s]; }
        __syncthreads();
    }
    if (tid == 0) {
        atomicAdd(&stats[oc],     rs[0]);
        atomicAdd(&stats[8 + oc], rq[0]);
    }
}

// stats -> per-channel (scale, shift):  yn = y*scale + shift
__global__ void bn_finalize(const float* __restrict__ stats,
                            const float* __restrict__ gamma, const float* __restrict__ beta,
                            float* __restrict__ ss, float invCnt, int C) {
    int c = threadIdx.x;
    if (c < C) {
        float mean = stats[c] * invCnt;
        float var  = stats[8 + c] * invCnt - mean * mean;
        float rstd = rsqrtf(var + BN_EPS);
        float sc   = gamma[c] * rstd;
        ss[c]     = sc;
        ss[8 + c] = beta[c] - mean * sc;
    }
}

// -------------------------- fused BN-apply + LIF (recurrence over T) + maxpool2
// One thread per pooled output location; 4 membrane potentials live in registers.
// toA=1: write f16 spikes directly into the WMMA A matrix [TB,1216] at c*HpWp+ph*Wp+pw.
__global__ void lif_pool(const float* __restrict__ y, const float* __restrict__ ss,
                         float* __restrict__ sOut, _Float16* __restrict__ aOut,
                         int B, int T, int C, int H, int W, int toA) {
    int Hp = H >> 1, Wp = W >> 1;
    int total = B * C * Hp * Wp;
    int idx = blockIdx.x * blockDim.x + threadIdx.x;
    if (idx >= total) return;
    int pw = idx % Wp; int r = idx / Wp;
    int ph = r % Hp;   r /= Hp;
    int c  = r % C;    int b = r / C;
    float scale = ss[c], shift = ss[8 + c];
    int HW = H * W;
    float v00 = 0.f, v01 = 0.f, v10 = 0.f, v11 = 0.f;
    for (int t = 0; t < T; ++t) {
        size_t base = ((size_t)(t * B + b) * C + c) * HW + (size_t)(2 * ph) * W + 2 * pw;
        float y00 = fmaf(y[base],         scale, shift);
        float y01 = fmaf(y[base + 1],     scale, shift);
        float y10 = fmaf(y[base + W],     scale, shift);
        float y11 = fmaf(y[base + W + 1], scale, shift);
        // v = v + (x - v)/2 ; spike = v>=1 ; hard reset
        v00 = 0.5f * (v00 + y00); float s00 = v00 >= V_TH ? 1.f : 0.f; v00 *= (1.f - s00);
        v01 = 0.5f * (v01 + y01); float s01 = v01 >= V_TH ? 1.f : 0.f; v01 *= (1.f - s01);
        v10 = 0.5f * (v10 + y10); float s10 = v10 >= V_TH ? 1.f : 0.f; v10 *= (1.f - s10);
        v11 = 0.5f * (v11 + y11); float s11 = v11 >= V_TH ? 1.f : 0.f; v11 *= (1.f - s11);
        float sm = fmaxf(fmaxf(s00, s01), fmaxf(s10, s11));
        if (toA) {
            aOut[(size_t)(t * B + b) * 1216 + c * (Hp * Wp) + ph * Wp + pw] = (_Float16)sm;
        } else {
            sOut[((size_t)(t * B + b) * C + c) * (Hp * Wp) + ph * Wp + pw] = sm;
        }
    }
}

// ------------------------------------------- FC1: [512,1216]x[1216,112] via WMMA
// A: spikes f16 row-major [TB,1216]; Bt: fw1 padded f16 [112,1216] (row = N, contiguous K)
// A frag (16x32 f16): lane m = L&15; K halves = hi*16 .. hi*16+15
// B frag (32x16 f16): lane n = L&15; K halves = hi*16 .. hi*16+15
// C/D (16x16 f32):    lane n = L&15; VGPR r -> row m0 + r + hi*8
__global__ void fc1_wmma(const _Float16* __restrict__ A, const _Float16* __restrict__ Bt,
                         float* __restrict__ out) {
    const int K = 1216;
    int wave = threadIdx.x >> 5;
    int lane = threadIdx.x & 31;
    int tile = blockIdx.x * 8 + wave;     // 28 blocks * 8 waves = 224 tiles (32 x 7)
    int tn = tile % 7, tm = tile / 7;
    int m0 = tm * 16, n0 = tn * 16;
    int mr = lane & 15;
    int hi = lane >> 4;
    const _Float16* aRow = A  + (size_t)(m0 + mr) * K + hi * 16;
    const _Float16* bRow = Bt + (size_t)(n0 + mr) * K + hi * 16;
    v8f acc = {};
    for (int k0 = 0; k0 < K; k0 += 32) {
        __builtin_prefetch((const void*)(aRow + k0 + 128), 0, 1);
        __builtin_prefetch((const void*)(bRow + k0 + 128), 0, 1);
        v8h alo = *(const v8h*)(aRow + k0);
        v8h ahi = *(const v8h*)(aRow + k0 + 8);
        v16h a;
        #pragma unroll
        for (int j = 0; j < 8; ++j) { a[j] = alo[j]; a[8 + j] = ahi[j]; }
        v16h bf = *(const v16h*)(bRow + k0);
        acc = __builtin_amdgcn_wmma_f32_16x16x32_f16(
            /*neg_a=*/false, a, /*neg_b=*/false, bf,
            /*c_mod=*/(short)0, acc, /*reuse_a=*/false, /*reuse_b=*/false);
    }
    int col = n0 + mr;
    if (col < 100) {
        #pragma unroll
        for (int r = 0; r < 8; ++r) {
            int row = m0 + r + hi * 8;
            out[(size_t)row * 100 + col] = acc[r];
        }
    }
}

// ---------------------------------- head: LIF(fc1) -> FC2 -> LIF, one block per b
__global__ void head_kernel(const float* __restrict__ pre, const float* __restrict__ fw2,
                            float* __restrict__ out, int B, int T) {
    int b = blockIdx.x;
    int tid = threadIdx.x;
    __shared__ float sp[100];
    float v1 = 0.f, v2 = 0.f;
    for (int t = 0; t < T; ++t) {
        if (tid < 100) {
            float x = pre[(size_t)(t * B + b) * 100 + tid];
            v1 = 0.5f * (v1 + x);
            float s = v1 >= V_TH ? 1.f : 0.f;
            v1 *= (1.f - s);
            sp[tid] = s;
        }
        __syncthreads();
        if (tid < 2) {
            float acc = 0.f;
            #pragma unroll 4
            for (int i = 0; i < 100; ++i) acc = fmaf(sp[i], fw2[tid * 100 + i], acc);
            v2 = 0.5f * (v2 + acc);
            float s2 = v2 >= V_TH ? 1.f : 0.f;
            v2 *= (1.f - s2);
            out[(size_t)(t * B + b) * 2 + tid] = s2;
        }
        __syncthreads();
    }
}

// ------------------------------------------------------------------- driver
extern "C" void kernel_launch(void* const* d_in, const int* in_sizes, int n_in,
                              void* d_out, int out_size, void* d_ws, size_t ws_size,
                              hipStream_t stream) {
    (void)in_sizes; (void)n_in; (void)out_size; (void)ws_size;
    const float* x   = (const float*)d_in[0];
    const float* w1  = (const float*)d_in[1];
    const float* g1  = (const float*)d_in[2];
    const float* b1  = (const float*)d_in[3];
    const float* w2  = (const float*)d_in[4];
    const float* g2  = (const float*)d_in[5];
    const float* b2  = (const float*)d_in[6];
    const float* w3  = (const float*)d_in[7];
    const float* g3  = (const float*)d_in[8];
    const float* b3  = (const float*)d_in[9];
    const float* fw1 = (const float*)d_in[10];
    const float* fw2 = (const float*)d_in[11];
    float* out = (float*)d_out;

    const int T = 16, B = 32, N = T * B, C = 6;

    // workspace carve-up (yBuf reused for y1/y2/y3; sBuf for s1/s2)
    char*  wsb   = (char*)d_ws;
    float* stats = (float*)wsb;                 // 3 x 16 f32 (sum[c], sumsq at +8)
    float* ss    = stats + 48;                  // 3 x 16 f32 (scale[c], shift at +8)
    float* yBuf  = (float*)(wsb + 1024);
    size_t ySz   = (size_t)N * C * 32 * 400;    // 157.3 MB
    float* sBuf  = yBuf + ySz;
    size_t sSz   = (size_t)N * C * 16 * 200;    // 39.3 MB
    float* fc1   = sBuf + sSz;                  // [512,100] f32
    _Float16* Ah = (_Float16*)(fc1 + (size_t)N * 100);   // [512,1216] f16 spikes
    _Float16* Bh = Ah + (size_t)N * 1216;                // [112,1216] f16 weights

    // init: stats zero, A matrix zero (covers K padding), pack fw1
    fill_zero_f32<<<1, 64, 0, stream>>>(stats, 48);
    {
        int nA = N * 1216;
        fill_zero_f16<<<(nA + 255) / 256, 256, 0, stream>>>(Ah, nA);
    }
    pack_fw1<<<(112 * 1216 + 255) / 256, 256, 0, stream>>>(fw1, Bh);

    // ---- block 1: conv(1->6) on 32x400 ----
    {
        int H = 32, W = 400, bpp = (H * W + 255) / 256;
        conv3x3_stats<1><<<N * C * bpp, 256, 0, stream>>>(x, w1, yBuf, stats + 0, N, C, H, W, bpp);
        bn_finalize<<<1, 32, 0, stream>>>(stats + 0, g1, b1, ss + 0, 1.f / ((float)N * H * W), C);
        int tot = B * C * (H / 2) * (W / 2);
        lif_pool<<<(tot + 255) / 256, 256, 0, stream>>>(yBuf, ss + 0, sBuf, nullptr, B, T, C, H, W, 0);
    }
    // ---- block 2: conv(6->6) on 16x200 ----
    {
        int H = 16, W = 200, bpp = (H * W + 255) / 256;
        conv3x3_stats<6><<<N * C * bpp, 256, 0, stream>>>(sBuf, w2, yBuf, stats + 16, N, C, H, W, bpp);
        bn_finalize<<<1, 32, 0, stream>>>(stats + 16, g2, b2, ss + 16, 1.f / ((float)N * H * W), C);
        int tot = B * C * (H / 2) * (W / 2);
        lif_pool<<<(tot + 255) / 256, 256, 0, stream>>>(yBuf, ss + 16, sBuf, nullptr, B, T, C, H, W, 0);
    }
    // ---- block 3: conv(6->6) on 8x100, pooled spikes go straight into A (f16) ----
    {
        int H = 8, W = 100, bpp = (H * W + 255) / 256;
        conv3x3_stats<6><<<N * C * bpp, 256, 0, stream>>>(sBuf, w3, yBuf, stats + 32, N, C, H, W, bpp);
        bn_finalize<<<1, 32, 0, stream>>>(stats + 32, g3, b3, ss + 32, 1.f / ((float)N * H * W), C);
        int tot = B * C * 4 * 50;
        lif_pool<<<(tot + 255) / 256, 256, 0, stream>>>(yBuf, ss + 32, nullptr, Ah, B, T, C, H, W, 1);
    }
    // ---- FC1 GEMM 512x100x1216 via v_wmma_f32_16x16x32_f16 (spikes are exact in f16) ----
    fc1_wmma<<<28, 256, 0, stream>>>(Ah, Bh, fc1);
    // ---- LIF -> FC2 -> LIF ----
    head_kernel<<<B, 128, 0, stream>>>(fc1, fw2, out, B, T);
}
